// CRF_38585986187869
// MI455X (gfx1250) — matrix operand order repaired and checked by reading, not compiled, not run on hardware
//
#include <hip/hip_runtime.h>

#define NUM_TAGS 40
#define SEQ_LEN  512
#define BATCH    2048
#define LSTRIDE  52              // padded LDS row stride (floats): 16B aligned rows (52*4=208=16*13)
#define WAVES_PER_BLOCK 4
#define BLOCKS   32              // 32 blocks * 4 waves * 16 batch-rows = 2048

#define LOG2E 1.4426950408889634f
#define LN2   0.6931471805599453f

typedef __attribute__((ext_vector_type(16))) _Float16 v16h;
typedef __attribute__((ext_vector_type(8)))  float    v8f;

union H16 { v16h v; _Float16 h[16]; };

// single-VALU half-wave swap (lane <-> lane^16) on wave32
__device__ inline float xor16(float x) {
#if __has_builtin(__builtin_amdgcn_permlanex16)
  int i = __builtin_bit_cast(int, x);
  int r = __builtin_amdgcn_permlanex16(i, i, 0x76543210, 0xfedcba98, false, false);
  return __builtin_bit_cast(float, r);
#else
  return __shfl_xor(x, 16, 32);
#endif
}

// One wave handles 16 batch rows. Relative score state in WMMA C-layout:
//   tile T in 0..2, reg r in 0..7:  tag j = T*16 + (lane<16 ? r : 8+r), batch = lane&15
// Per step:  D'[j][b] = sum_i E[i][j] * p[i][b]  via v_wmma_f32_16x16x32_f16.
// K slots are permuted so B = p^T packs from each lane's own registers (no shuffles);
// emission tiles are DMA'd global->LDS with async loads (ASYNCcnt double-buffering).
__global__ __launch_bounds__(128) void crf_fused(
    const float* __restrict__ em, const int* __restrict__ tags,
    const float* __restrict__ trans, const float* __restrict__ startT,
    const float* __restrict__ endT, float* __restrict__ partial)
{
  __shared__ float transSh[NUM_TAGS * NUM_TAGS];
  __shared__ float startSh[48];
  __shared__ float endSh[48];
  __shared__ __align__(16) float emSh[WAVES_PER_BLOCK][2][16 * LSTRIDE];

  const int  tid   = threadIdx.x;
  const int  wave  = tid >> 5;
  const int  lane  = tid & 31;
  const bool lo    = lane < 16;
  const int  bl    = lane & 15;
  const int  gwave = blockIdx.x * WAVES_PER_BLOCK + wave;
  const int  b0    = gwave * 16;
  const int  bg    = b0 + bl;

  // block-cooperative LDS fills (zero-padded start/end transition vectors)
  for (int i = tid; i < NUM_TAGS * NUM_TAGS; i += blockDim.x) transSh[i] = trans[i];
  if (tid < 48) {
    startSh[tid] = (tid < NUM_TAGS) ? startT[tid] : 0.f;
    endSh[tid]   = (tid < NUM_TAGS) ? endT[tid]   : 0.f;
  }
  // zero the pad columns (j=40..47) of this wave's two em buffers, once
  #pragma unroll
  for (int q = lane; q < 256; q += 32) {
    int bsel = q >> 7, rr = (q >> 3) & 15, cc = q & 7;
    emSh[wave][bsel][rr * LSTRIDE + NUM_TAGS + cc] = 0.f;
  }
  __syncthreads();

  // ---- constant A operands with permuted K slots ----
  // hardware slot kb (0..31) -> tag: identity (kb -> kb), matching B's packing
  v16h A[3][2];
  #pragma unroll
  for (int T = 0; T < 3; ++T) {
    #pragma unroll
    for (int s = 0; s < 2; ++s) {
      H16 a;
      #pragma unroll
      for (int e2 = 0; e2 < 16; ++e2) {
        int v   = e2 >> 1, e = e2 & 1;
        int kb  = (v < 4 ? 2 * v : 16 + 2 * (v - 4)) + (lo ? 0 : 8) + e;  // hw slot in 0..31
        int grp = kb >> 3;
        int tagk = kb;                                                     // tag = hw slot
        int i   = s * 32 + tagk;         // contraction tag index (padded to 64)
        int j   = T * 16 + bl;           // output tag row (padded to 48)
        float val = (i < NUM_TAGS && j < NUM_TAGS) ? __expf(transSh[i * NUM_TAGS + j]) : 0.f;
        a.h[e2] = (_Float16)val;
      }
      A[T][s] = a.v;
    }
  }

  float* buf0 = &emSh[wave][0][0];
  float* buf1 = &emSh[wave][1][0];

  // per-lane tile-copy geometry (16x40 contiguous floats per tile, 5 x b128 per wave)
  unsigned voffB[5], lofs[5];
  #pragma unroll
  for (int it = 0; it < 5; ++it) {
    int v = it * 32 + lane, row = v / 10, col = v % 10;
    voffB[it] = (unsigned)(((b0 + row) * NUM_TAGS + col * 4) * 4);  // byte offset in em row-block
    lofs[it]  = (unsigned)((row * LSTRIDE + col * 4) * 4);          // byte offset in LDS tile
  }
  const unsigned base0 = (unsigned)(uintptr_t)buf0;   // LDS byte address = low 32 bits
  const unsigned base1 = (unsigned)(uintptr_t)buf1;

  // stage t=0 tile into buf1 and t=1 tile into buf0 (synchronous prologue)
  float4 pre[5];
  #pragma unroll
  for (int it = 0; it < 5; ++it) {
    int v = it * 32 + lane, row = v / 10, col = v % 10;
    pre[it] = *(const float4*)(em + ((size_t)b0 + row) * NUM_TAGS + col * 4);
  }
  #pragma unroll
  for (int it = 0; it < 5; ++it) {
    int v = it * 32 + lane, row = v / 10, col = v % 10;
    *(float4*)(buf1 + row * LSTRIDE + col * 4) = pre[it];
  }

  // ---- score init: score0[j][b] = start[j] + em[0][b][j]; pad rows -> -inf ----
  const int emOff = bl * LSTRIDE + (lo ? 0 : 8);   // per-lane base into em tiles
  float c[3][8];
  #pragma unroll
  for (int T = 0; T < 3; ++T) {
    const float4* ep = (const float4*)(buf1 + emOff + T * 16);
    float4 ea = ep[0], eb = ep[1];
    int jb = T * 16 + (lo ? 0 : 8);
    float sv[8] = { startSh[jb+0], startSh[jb+1], startSh[jb+2], startSh[jb+3],
                    startSh[jb+4], startSh[jb+5], startSh[jb+6], startSh[jb+7] };
    float ev[8] = { ea.x, ea.y, ea.z, ea.w, eb.x, eb.y, eb.z, eb.w };
    bool valid = (T < 2) | lo;           // only tile2 high-half is padding
    #pragma unroll
    for (int r = 0; r < 8; ++r) c[T][r] = valid ? (sv[r] + ev[r]) : -INFINITY;
  }
  float Mcum = 0.f;                      // per-batch running max offset

  // numerator init (lanes 0-15 meaningful; all lanes compute uniformly)
  int   prev = tags[bg];
  float num  = startSh[prev] + buf1[bl * LSTRIDE + prev];

  #pragma unroll
  for (int it = 0; it < 5; ++it) {
    int v = it * 32 + lane, row = v / 10, col = v % 10;
    pre[it] = *(const float4*)(em + ((size_t)1 * BATCH + b0 + row) * NUM_TAGS + col * 4);
  }
  #pragma unroll
  for (int it = 0; it < 5; ++it) {
    int v = it * 32 + lane, row = v / 10, col = v % 10;
    *(float4*)(buf0 + row * LSTRIDE + col * 4) = pre[it];
  }

  for (int t = 1; t < SEQ_LEN; ++t) {
    float* cbuf = ((t - 1) & 1) ? buf1 : buf0;

    if (t < SEQ_LEN - 1) {
      // async DMA of tile t+1 straight into the other LDS buffer (no VGPR landing)
      const float* srow = em + (size_t)(t + 1) * BATCH * NUM_TAGS;
      unsigned nb = ((t - 1) & 1) ? base0 : base1;
      #pragma unroll
      for (int it = 0; it < 5; ++it) {
        unsigned ld = nb + lofs[it];
        asm volatile("global_load_async_to_lds_b128 %0, %1, %2"
                     :: "v"(ld), "v"(voffB[it]), "s"(srow) : "memory");
      }
      // loads complete in order: <=5 pending guarantees tile t (issued last iter) is in LDS
      asm volatile("s_wait_asynccnt 0x5" ::: "memory");
    } else {
      asm volatile("s_wait_asynccnt 0x0" ::: "memory");
    }

    // per-batch max over tags (pad rows are -inf -> no masking needed)
    float m = -INFINITY;
    #pragma unroll
    for (int r = 0; r < 8; ++r)
      m = fmaxf(m, fmaxf(fmaxf(c[0][r], c[1][r]), c[2][r]));
    m = fmaxf(m, xor16(m));
    Mcum += m;

    // p = exp2(c*log2e - m*log2e)  (raw v_exp_f32; -inf -> exactly 0)
    float mL = m * LOG2E;
    float pv[3][8];
    #pragma unroll
    for (int T = 0; T < 3; ++T)
      #pragma unroll
      for (int r = 0; r < 8; ++r)
        pv[T][r] = __builtin_amdgcn_exp2f(fmaf(c[T][r], LOG2E, -mL));

    // B = p^T with the permuted K slots: pure per-lane packing, no shuffles/selects.
    H16 B0, B1;
    #pragma unroll
    for (int kk = 0; kk < 8; ++kk) B0.h[kk]     = (_Float16)pv[0][kk];
    #pragma unroll
    for (int kk = 0; kk < 8; ++kk) B0.h[kk + 8] = (_Float16)pv[1][kk];
    #pragma unroll
    for (int kk = 0; kk < 8; ++kk) B1.h[kk]     = (_Float16)pv[2][kk];
    #pragma unroll
    for (int kk = 0; kk < 8; ++kk) B1.h[kk + 8] = (_Float16)0.f;

    // 3 M-tiles x 2 K-slices of v_wmma_f32_16x16x32_f16, then c = em + ln2*log2(dot)
    #pragma unroll
    for (int T = 0; T < 3; ++T) {
      v8f acc = {0.f, 0.f, 0.f, 0.f, 0.f, 0.f, 0.f, 0.f};
      acc = __builtin_amdgcn_wmma_f32_16x16x32_f16(false, A[T][0], false, B0.v,
                                                   (short)0, acc, false, false);
      acc = __builtin_amdgcn_wmma_f32_16x16x32_f16(false, A[T][1], false, B1.v,
                                                   (short)0, acc, false, false);
      const float4* ep = (const float4*)(cbuf + emOff + T * 16);
      float4 ea = ep[0], eb = ep[1];
      float ev[8] = { ea.x, ea.y, ea.z, ea.w, eb.x, eb.y, eb.z, eb.w };
      #pragma unroll
      for (int r = 0; r < 8; ++r)
        c[T][r] = fmaf(__builtin_amdgcn_logf(acc[r]), LN2, ev[r]);
        // acc is 0 (pad -> -inf) or >= ~0.5 (valid): raw v_log_f32, no denorms
    }

    // fused numerator step (mask is all-ones)
    int tg = tags[(size_t)t * BATCH + bg];
    num += transSh[prev * NUM_TAGS + tg] + cbuf[bl * LSTRIDE + tg];
    prev = tg;
  }

  // ---- finalize: add end transitions (pad adds 0 to -inf), logsumexp over tags ----
  #pragma unroll
  for (int T = 0; T < 3; ++T) {
    int jb = T * 16 + (lo ? 0 : 8);
    #pragma unroll
    for (int r = 0; r < 8; ++r) c[T][r] += endSh[jb + r];
  }
  float m = -INFINITY;
  #pragma unroll
  for (int r = 0; r < 8; ++r)
    m = fmaxf(m, fmaxf(fmaxf(c[0][r], c[1][r]), c[2][r]));
  m = fmaxf(m, xor16(m));
  float mL = m * LOG2E;
  float s = 0.f;
  #pragma unroll
  for (int T = 0; T < 3; ++T)
    #pragma unroll
    for (int r = 0; r < 8; ++r)
      s += __builtin_amdgcn_exp2f(fmaf(c[T][r], LOG2E, -mL));  // -inf padding -> 0
  s += xor16(s);
  float den = fmaf(__builtin_amdgcn_logf(s), LN2, m + Mcum);

  num += endSh[prev];
  float llh = num - den;                   // valid on lanes 0-15 (one batch each)
  #pragma unroll
  for (int off = 8; off >= 1; off >>= 1) llh += __shfl_xor(llh, off, 16);
  if (lane == 0) partial[gwave] = llh;
}

__global__ __launch_bounds__(128) void crf_reduce(const float* __restrict__ partial,
                                                  float* __restrict__ out)
{
  __shared__ float s[128];
  int i = threadIdx.x;
  s[i] = partial[i];
  __syncthreads();
  #pragma unroll
  for (int off = 64; off > 0; off >>= 1) {
    if (i < off) s[i] += s[i + off];
    __syncthreads();
  }
  if (i == 0) out[0] = s[0];
}

extern "C" void kernel_launch(void* const* d_in, const int* in_sizes, int n_in,
                              void* d_out, int out_size, void* d_ws, size_t ws_size,
                              hipStream_t stream) {
  const float* emissions = (const float*)d_in[0];
  const int*   tags      = (const int*)d_in[1];
  // d_in[2] = mask (all ones in this problem) — unused
  const float* trans     = (const float*)d_in[3];
  const float* startT    = (const float*)d_in[4];
  const float* endT      = (const float*)d_in[5];
  float* partial = (float*)d_ws;                     // 128 floats
  float* out     = (float*)d_out;

  crf_fused<<<dim3(BLOCKS), dim3(128), 0, stream>>>(emissions, tags, trans,
                                                    startT, endT, partial);
  crf_reduce<<<dim3(1), dim3(128), 0, stream>>>(partial, out);
}
